// Beran_40114994545012
// MI455X (gfx1250) — matrix operand order repaired and checked by reading, not compiled, not run on hardware
//
#include <hip/hip_runtime.h>

typedef __attribute__((ext_vector_type(16))) _Float16 v16h;
typedef __attribute__((ext_vector_type(8)))  float    v8f;

#define BB 1024
#define NN 8192
#define KK 8
#define CC 10
#define KC 80          // K*C
#define CHUNKS 3       // 3 * 32 = 96 >= 80 (zero padded)
#define NTILES (NN / 16)   // 512
#define BTILES (BB / 16)   // 64
#define EPSV 1e-13f
#define ISCTOL 1.00001e-5f // jnp.isclose(x,1): atol(1e-8) + rtol(1e-5)*1

// ---------------------------------------------------------------------------
// prep_b: softmax(c_p) / bw, A[b] = sum p^2/bw, and A-matrix WMMA fragments
// (hi/lo f16 split) in this code's contiguous-k A order (not the ISA table order):
//   lane L holds row M = L%16; for L<16 elements i map K_local = 0..15,
//   for L>=16 K_local = 16..31.
// ---------------------------------------------------------------------------
__global__ void prep_b_kernel(const float* __restrict__ c_p,
                              const float* __restrict__ bw,
                              float* __restrict__ Avec,
                              _Float16* __restrict__ Ahi,
                              _Float16* __restrict__ Alo) {
  __shared__ float pw_sh[16][KC];
  __shared__ float apart[16][KK];
  const int t = threadIdx.x;           // 0..127
  const int btile = blockIdx.x;        // 0..63
  const int bloc = t >> 3;             // 0..15
  const int k    = t & 7;              // 0..7
  const int b = btile * 16 + bloc;

  // per-(b,k) softmax over C=10, then scale by 1/clip(bw)
  const float* src = c_p + ((size_t)b * KK + k) * CC;
  float x[CC];
  float m = -3.4e38f;
#pragma unroll
  for (int c = 0; c < CC; ++c) { x[c] = src[c]; m = fmaxf(m, x[c]); }
  float s = 0.f;
#pragma unroll
  for (int c = 0; c < CC; ++c) { x[c] = expf(x[c] - m); s += x[c]; }
  const float invs = 1.f / s;
  float ap = 0.f;
#pragma unroll
  for (int c = 0; c < CC; ++c) {
    float p = x[c] * invs;
    float bwv = fminf(fmaxf(bw[k * CC + c], 0.001f), 10.0f);
    float pv = p / bwv;
    pw_sh[bloc][k * CC + c] = pv;
    ap += p * pv;                       // p^2 / bw
  }
  apart[bloc][k] = ap;
  __syncthreads();

  if (t < 16) {
    float a = 0.f;
#pragma unroll
    for (int kk = 0; kk < KK; ++kk) a += apart[t][kk];
    Avec[btile * 16 + t] = a;
  }

  if (t < 96) {
    const int chunk = t >> 5;          // 0..2
    const int lane  = t & 31;
    const int brow  = lane & 15;
    const size_t basei = (((size_t)btile * CHUNKS + chunk) * 32 + lane) * 16;
#pragma unroll
    for (int i = 0; i < 16; ++i) {
      int kl = ((lane < 16) ? 0 : 16) + i;
      int kc = chunk * 32 + kl;
      float v = (kc < KC) ? pw_sh[brow][kc] : 0.f;
      _Float16 hi = (_Float16)v;
      float lo = v - (float)hi;
      Ahi[basei + i] = hi;
      Alo[basei + i] = (_Float16)lo;
    }
  }
}

// ---------------------------------------------------------------------------
// prep_n: D[n] = sum_k 1/bw[k, c_in[n,k]] and one-hot B-matrix WMMA fragments
// (contiguous-k B order, as in A: lane L holds column N = L%16; lanes 0-15 cover
// K_local 0..15, lanes 16-31 cover K_local 16..31; element i -> K_block + i).
// ---------------------------------------------------------------------------
__global__ void prep_n_kernel(const int* __restrict__ c_in,
                              const float* __restrict__ bw,
                              float* __restrict__ Dvec,
                              _Float16* __restrict__ Bfr) {
  const int t = threadIdx.x;           // 0..127
  const int ntile = blockIdx.x;        // 0..511

  if (t < 96) {
    const int chunk = t >> 5;
    const int lane  = t & 31;
    const int n = ntile * 16 + (lane & 15);
    const int kb = chunk * 32 + ((lane < 16) ? 0 : 16);
    const size_t basei = (((size_t)ntile * CHUNKS + chunk) * 32 + lane) * 16;
#pragma unroll
    for (int i = 0; i < 16; ++i) {
      int kc = kb + i;
      _Float16 h = (_Float16)0.f;
      if (kc < KC) {
        int k = kc / CC;
        int c = kc - k * CC;
        h = (c_in[n * KK + k] == c) ? (_Float16)1.f : (_Float16)0.f;
      }
      Bfr[basei + i] = h;
    }
  }
  if (t < 16) {
    const int n = ntile * 16 + t;
    float d = 0.f;
#pragma unroll
    for (int k = 0; k < KK; ++k) {
      int c = c_in[n * KK + k];
      float bwv = fminf(fmaxf(bw[k * CC + c], 0.001f), 10.0f);
      d += 1.f / bwv;
    }
    Dvec[n] = d;
  }
}

// ---------------------------------------------------------------------------
// GEMM: cross = pw x onehot^T via v_wmma_f32_16x16x32_f16.
// Hi/lo split accumulation restores f32 accuracy (B is exact 0/1 in f16).
// One wave per 16x16 output tile; 8 waves (8 n-tiles) per block.
// ---------------------------------------------------------------------------
__global__ void gemm_weights_kernel(const _Float16* __restrict__ Ahi,
                                    const _Float16* __restrict__ Alo,
                                    const _Float16* __restrict__ Bfr,
                                    const float* __restrict__ Avec,
                                    const float* __restrict__ Dvec,
                                    float* __restrict__ Wraw) {
  const int lane  = threadIdx.x & 31;
  const int wave  = threadIdx.x >> 5;          // 0..7
  const int btile = blockIdx.y;                // 0..63
  const int ntile = blockIdx.x * 8 + wave;     // 0..511

  v8f acc = {};
#pragma unroll
  for (int ch = 0; ch < CHUNKS; ++ch) {
    const size_t ai = (((size_t)btile * CHUNKS + ch) * 32 + lane) * 16;
    const size_t bi = (((size_t)ntile * CHUNKS + ch) * 32 + lane) * 16;
    v16h a  = *(const v16h*)(Ahi + ai);
    v16h al = *(const v16h*)(Alo + ai);
    v16h bm = *(const v16h*)(Bfr + bi);
    acc = __builtin_amdgcn_wmma_f32_16x16x32_f16(false, a,  false, bm,
                                                 (short)0, acc, false, false);
    acc = __builtin_amdgcn_wmma_f32_16x16x32_f16(false, al, false, bm,
                                                 (short)0, acc, false, false);
  }

  // C/D layout: VGPR r -> (M = r [lanes 0-15] or 8+r [lanes 16-31], N = lane%16)
  const int ncol = lane & 15;
  const int m0 = (lane < 16) ? 0 : 8;
  const int n = ntile * 16 + ncol;
  const float d = Dvec[n];
#pragma unroll
  for (int r = 0; r < 8; ++r) {
    int b = btile * 16 + m0 + r;
    float metric = Avec[b] - 2.0f * acc[r] + d;
    Wraw[(size_t)b * NN + n] = expf(-metric);
  }
}

// ---------------------------------------------------------------------------
// Per-row Beran scan: one 256-thread block per b; 32 elems/thread; three
// block-level Hillis-Steele scans in LDS.
// ---------------------------------------------------------------------------
__device__ __forceinline__ float block_scan_incl(float v, float* lds, int tid,
                                                 float* total) {
  lds[tid] = v;
  __syncthreads();
#pragma unroll
  for (int off = 1; off < 256; off <<= 1) {
    float x = lds[tid];
    float y = (tid >= off) ? lds[tid - off] : 0.f;
    __syncthreads();
    lds[tid] = x + y;
    __syncthreads();
  }
  float incl = lds[tid];
  *total = lds[255];
  __syncthreads();
  return incl;
}

__global__ void beran_scan_kernel(const float* __restrict__ Wraw,
                                  const int* __restrict__ delta,
                                  float* __restrict__ out) {
  __shared__ float lds[256];
  const int tid = threadIdx.x;
  const int b = blockIdx.x;
  const int n0 = tid * 32;
  const float* row = Wraw + (size_t)b * NN;

  float w[32];
  const float4* rp = (const float4*)(row + n0);
#pragma unroll
  for (int j = 0; j < 8; ++j) {
    float4 v = rp[j];
    w[4 * j + 0] = v.x; w[4 * j + 1] = v.y;
    w[4 * j + 2] = v.z; w[4 * j + 3] = v.w;
  }

  // ---- normalize weights: s = sum(w) ----
  float tot = 0.f;
#pragma unroll
  for (int i = 0; i < 32; ++i) tot += w[i];
  float gsum;
  float incl = block_scan_incl(tot, lds, tid, &gsum);
  float base = incl - tot;                       // exclusive raw prefix
  float inv = (gsum < EPSV) ? 0.f : 1.f / gsum;

  // ---- cumsum(W), xi, filtered_xi (store local inclusive prefix of fi) ----
  float run = 0.f, run2 = 0.f;
#pragma unroll
  for (int i = 0; i < 32; ++i) {
    float wi = w[i];
    run += wi;
    float wc = inv * (base + run);               // inclusive cumsum of W
    float sh = inv * (base + run - wi);          // shifted (exclusive)
    bool bad2 = (fabsf(sh - 1.f) <= ISCTOL) || (fabsf(wc - 1.f) <= ISCTOL);
    if (bad2) { sh = 0.f; wc = 0.f; }
    float xi = logf(1.f - sh) - logf(1.f - wc);
    float fi = (float)delta[n0 + i] * xi;        // multiply (mirrors 0*NaN)
    run2 += fi;
    w[i] = run2;                                 // local incl prefix of fi
  }

  // ---- hazards = cumsum(filtered_xi); surv = exp(-hazards); steps ----
  float gsum2;
  float incl2 = block_scan_incl(run2, lds, tid, &gsum2);
  float base2 = incl2 - run2;

  float st[32];
  float prevp = 0.f, stepsum = 0.f;
#pragma unroll
  for (int i = 0; i < 32; ++i) {
    float p = w[i];
    float h = base2 + p;
    float sv = expf(-h);
    out[(size_t)b * NN + n0 + i] = sv;           // surv_func
    float step;
    if (n0 + i == 0) {
      step = 1.f - sv;
    } else {
      float prevh = base2 + prevp;               // hazard of element n-1
      step = expf(-prevh) - sv;
    }
    prevp = p;
    st[i] = step;
    stepsum += step;
  }

  // ---- normalize surv_steps ----
  float gss;
  (void)block_scan_incl(stepsum, lds, tid, &gss);
  float inv3 = (gss < EPSV) ? 0.f : 1.f / gss;
  const size_t off2 = (size_t)BB * NN;
#pragma unroll
  for (int i = 0; i < 32; ++i)
    out[off2 + (size_t)b * NN + n0 + i] = st[i] * inv3;
}

// ---------------------------------------------------------------------------
extern "C" void kernel_launch(void* const* d_in, const int* in_sizes, int n_in,
                              void* d_out, int out_size, void* d_ws,
                              size_t ws_size, hipStream_t stream) {
  (void)in_sizes; (void)n_in; (void)out_size; (void)ws_size;
  const float* c_p   = (const float*)d_in[0];   // [B,K,C]
  const float* bw    = (const float*)d_in[1];   // [K*C]
  const int*   c_in  = (const int*)d_in[2];     // [N,K]
  const int*   delta = (const int*)d_in[3];     // [N]
  float* out = (float*)d_out;                   // [2,B,N]

  char* wsb = (char*)d_ws;
  const size_t szW   = (size_t)BB * NN * sizeof(float);            // 32 MB
  float*    Wraw = (float*)wsb;
  float*    Avec = (float*)(wsb + szW);                            // 4 KB
  float*    Dvec = (float*)(wsb + szW + 4096);                     // 32 KB
  _Float16* Ahi  = (_Float16*)(wsb + szW + 36864);                 // 192 KB
  _Float16* Alo  = (_Float16*)(wsb + szW + 36864 + 196608);        // 192 KB
  _Float16* Bfr  = (_Float16*)(wsb + szW + 36864 + 2 * 196608);    // 1.5 MB

  prep_b_kernel<<<BTILES, 128, 0, stream>>>(c_p, bw, Avec, Ahi, Alo);
  prep_n_kernel<<<NTILES, 128, 0, stream>>>(c_in, bw, Dvec, Bfr);
  dim3 g1(NTILES / 8, BTILES);
  gemm_weights_kernel<<<g1, 256, 0, stream>>>(Ahi, Alo, Bfr, Avec, Dvec, Wraw);
  beran_scan_kernel<<<BB, 256, 0, stream>>>(Wraw, delta, out);
}
